// Spatial_Descriptor_43748536877383
// MI455X (gfx1250) — matrix operand order repaired and checked by reading, not compiled, not run on hardware
//
#include <hip/hip_runtime.h>

#define FN_ 100000
#define VN_ 50000
#define BN_ 8

typedef _Float16 h8   __attribute__((ext_vector_type(8)));
typedef _Float16 h16  __attribute__((ext_vector_type(16)));
typedef float    f8v  __attribute__((ext_vector_type(8)));
typedef unsigned int u32x4 __attribute__((ext_vector_type(4)));
typedef int      i32x4 __attribute__((ext_vector_type(4)));
typedef int      i32x8 __attribute__((ext_vector_type(8)));

static __device__ inline h16 cat8(h8 lo, h8 hi) {
  return __builtin_shufflevector(lo, hi, 0,1,2,3,4,5,6,7,8,9,10,11,12,13,14,15);
}

static __device__ inline f8v zero8() {
  f8v z;
#pragma unroll
  for (int i = 0; i < 8; ++i) z[i] = 0.0f;
  return z;
}

// A operand: weights row-major [64][ldk] f16 in LDS.
// 16-bit A 16x32 layout: lanes 0-15 row M=lane K={0..7,16..23}; lanes 16-31 row M=lane-16 K={8..15,24..31}
static __device__ inline h16 loadA(const _Float16* W, int ldk, int mt, int kt, int lane) {
  int m  = mt * 16 + (lane & 15);
  int ho = (lane >> 4) * 8;
  const _Float16* p = W + m * ldk + kt * 32;
  h8 a0 = *(const h8*)(p + ho);
  h8 a1 = *(const h8*)(p + 16 + ho);
  return cat8(a0, a1);
}

// B operand: activations column-major [32][ldk] f16 in LDS; sub selects N group.
// 16-bit B 32x16 layout: lane holds column N=lane%16, K={0..7,16..23} (lanes 0-15) / {8..15,24..31} (lanes 16-31)
static __device__ inline h16 loadB(const _Float16* Xc, int ldk, int kt, int lane, int sub) {
  const _Float16* p = Xc + (sub * 16 + (lane & 15)) * ldk + kt * 32 + (lane >> 4) * 8;
  h8 b0 = *(const h8*)(p);
  h8 b1 = *(const h8*)(p + 16);
  return cat8(b0, b1);
}

static __device__ inline f8v wmma_f16(h16 a, h16 b, f8v c) {
  return __builtin_amdgcn_wmma_f32_16x16x32_f16(false, a, false, b, (short)0, c, false, false);
}

// D/C 16x16 f32 layout: VGPR r, lane l -> M = mt*16 + (l>=16 ? 8 : 0) + r, N = l%16
static __device__ inline void store_relu_h(_Float16* Hc, int ldk, int mt, int lane, int sub,
                                           f8v acc, const float* bias) {
  int n   = sub * 16 + (lane & 15);
  int ch0 = mt * 16 + (lane >> 4) * 8;
  h8 hv;
#pragma unroll
  for (int r = 0; r < 8; ++r) hv[r] = (_Float16)fmaxf(acc[r] + bias[ch0 + r], 0.0f);
  *(h8*)(Hc + n * ldk + ch0) = hv;
}

// intra-wave LDS producer->consumer: wave32 is lockstep, only DS completion is needed
static __device__ inline void wave_lds_fence() {
  asm volatile("s_wait_dscnt 0x0" ::: "memory");
}

struct PrepArgs {
  const float *cw1, *cb1, *c1g, *c1bt, *c1m, *c1v;
  const float *cw2, *cb2, *c2g, *c2bt, *c2m, *c2v;
  const float *vw1, *vb1, *v1g, *v1bt, *v1m, *v1v;
  const float *vw2, *vb2, *v2g, *v2bt, *v2m, *v2v;
  const float *sw,  *sb,  *s1g, *s1bt, *s1m, *s1v;
};

// Fold eval-BN into conv weights/biases, cast to f16, pad layer-1 K to 32,
// fold the /3 vfeat average into the last 64 columns of sw.
// ws f16 layout (halves):  [0,2048) vW1p[64][32]  [2048,4096) cW1p[64][32]
//   [4096,8192) vW2[64][64]  [8192,12288) cW2[64][64]  [12288,20480) sW[64][128]
// ws f32 biases at byte 40960: v1,c1,v2,c2,s (5 x 64 floats) -> total 42240 B
__global__ void prep_kernel(PrepArgs p, _Float16* wH, float* bF) {
  int o = threadIdx.x;
  if (o >= 64) return;
  const float eps = 1e-5f;
  {
    float s = p.v1g[o] * rsqrtf(p.v1v[o] + eps);
    for (int k = 0; k < 32; ++k)
      wH[o * 32 + k] = (_Float16)((k < 6) ? p.vw1[o * 6 + k] * s : 0.0f);
    bF[o] = s * (p.vb1[o] - p.v1m[o]) + p.v1bt[o];
  }
  {
    float s = p.c1g[o] * rsqrtf(p.c1v[o] + eps);
    for (int k = 0; k < 32; ++k)
      wH[2048 + o * 32 + k] = (_Float16)((k < 3) ? p.cw1[o * 3 + k] * s : 0.0f);
    bF[64 + o] = s * (p.cb1[o] - p.c1m[o]) + p.c1bt[o];
  }
  {
    float s = p.v2g[o] * rsqrtf(p.v2v[o] + eps);
    for (int k = 0; k < 64; ++k)
      wH[4096 + o * 64 + k] = (_Float16)(p.vw2[o * 64 + k] * s);
    bF[128 + o] = s * (p.vb2[o] - p.v2m[o]) + p.v2bt[o];
  }
  {
    float s = p.c2g[o] * rsqrtf(p.c2v[o] + eps);
    for (int k = 0; k < 64; ++k)
      wH[8192 + o * 64 + k] = (_Float16)(p.cw2[o * 64 + k] * s);
    bF[192 + o] = s * (p.cb2[o] - p.c2m[o]) + p.c2bt[o];
  }
  {
    float s = p.s1g[o] * rsqrtf(p.s1v[o] + eps);
    for (int k = 0; k < 64; ++k)
      wH[12288 + o * 128 + k] = (_Float16)(p.sw[o * 128 + k] * s);
    for (int k = 64; k < 128; ++k)
      wH[12288 + o * 128 + k] = (_Float16)(p.sw[o * 128 + k] * s * (1.0f / 3.0f));
    bF[256 + o] = s * (p.sb[o] - p.s1m[o]) + p.s1bt[o];
  }
}

// LDS: [0,40960) f16 weights; [40960,42240) f32 biases;
// per-wave scratch @42240 + wave*24576:
//   X = 4 x [32][32] f16 (8KB), H = 4 x [32][64] f16 (16KB); S [32][128] reuses X.
// Each wave owns a 32-position macro-tile (two WMMA column tiles) so every
// A (weight) tile fetched from LDS feeds two v_wmma issues.
__global__ __launch_bounds__(128) void spatial_desc_main(
    const float* __restrict__ centers, const float* __restrict__ verts,
    const int* __restrict__ faces, const uint4* __restrict__ wsrc,
    float* __restrict__ out) {
  extern __shared__ char lds[];
  const int tid = threadIdx.x, wave = tid >> 5, lane = tid & 31;

  // ---- stage folded weights + biases into LDS (42240 B) via Tensor Data Mover ----
#if __has_builtin(__builtin_amdgcn_tensor_load_to_lds)
  if (wave == 0) {  // one TDM descriptor: 1-D tile, 5280 x 8B elements
    unsigned long long ga = (unsigned long long)(const void*)wsrc;
    u32x4 g0;
    g0[0] = 1u;                                                // count=1, user mode
    g0[1] = 0u;                                                // lds_addr = 0
    g0[2] = (unsigned)(ga & 0xffffffffu);                      // global_addr[31:0]
    g0[3] = (unsigned)((ga >> 32) & 0x01ffffffu) | (2u << 30); // addr[56:32] | type=2
    i32x8 g1;
    g1[0] = 3 << 16;             // data_size = 8B, no multicast/pad/iterate
    g1[1] = (int)(5280u << 16);  // tensor_dim0 = 5280 (low 16 bits)
    g1[2] = 1 << 16;             // tensor_dim0 hi=0 | tensor_dim1 = 1
    g1[3] = (int)(5280u << 16);  // tile_dim0 = 5280
    g1[4] = 1;                   // tile_dim1 = 1, tile_dim2 = 0
    g1[5] = 5280;                // tensor_dim0_stride
    g1[6] = 0; g1[7] = 0;
    i32x4 gz4; gz4[0] = 0; gz4[1] = 0; gz4[2] = 0; gz4[3] = 0;
    i32x8 gz8;
#pragma unroll
    for (int j = 0; j < 8; ++j) gz8[j] = 0;
    // clang-23 / therock-10.0 form: (g0, g1, g2, g3, g4, cpol)
    __builtin_amdgcn_tensor_load_to_lds(g0, g1, gz4, gz4, gz8, 0);
    __builtin_amdgcn_s_wait_tensorcnt(0);
  }
#else
  {
    uint4* dst = (uint4*)lds;
    for (int i = tid; i < 2640; i += 128) dst[i] = wsrc[i];
  }
#endif
  __syncthreads();

  _Float16*    Wlds = (_Float16*)lds;
  const float* Blds = (const float*)(lds + 40960);
  _Float16*    X    = (_Float16*)(lds + 42240 + wave * 24576);  // 4096 halves
  _Float16*    H    = X + 4096;                                 // 8192 halves

  const int tile = blockIdx.x * 4 + wave;  // 25000 macro-tiles total
  const int base = tile * 32;
  const int b    = base / FN_;  // FN_ % 32 == 0: tile never straddles batch
  const int f0   = base % FN_;

  // ---- Stage 0: gather vertices/centers, build layer-1 B operands (all 32 lanes) ----
  {
    const int f = f0 + lane;
    const int* fc = faces + ((size_t)b * FN_ + f) * 3;
    int i0 = fc[0], i1 = fc[1], i2 = fc[2];
    const float* vb = verts + (size_t)b * VN_ * 3;
    const float* p0 = vb + (size_t)i0 * 3;
    const float* p1 = vb + (size_t)i1 * 3;
    const float* p2 = vb + (size_t)i2 * 3;
    float v0x = p0[0], v0y = p0[1], v0z = p0[2];
    float v1x = p1[0], v1y = p1[1], v1z = p1[2];
    float v2x = p2[0], v2y = p2[1], v2z = p2[2];
    const float* cb = centers + (size_t)b * 3 * FN_ + f;
    float c0 = cb[0], c1 = cb[FN_], c2 = cb[2 * FN_];

    _Float16 col[32];
#pragma unroll
    for (int j = 0; j < 32; ++j) col[j] = (_Float16)0.0f;

    // branch 0: xc (K=3 used)
    col[0] = (_Float16)c0; col[1] = (_Float16)c1; col[2] = (_Float16)c2;
#pragma unroll
    for (int j = 0; j < 4; ++j) *(h8*)(X + 0 * 1024 + lane * 32 + j * 8) = *(h8*)(col + j * 8);
    // branch 1: x01 = (v0, v1)
    col[0] = (_Float16)v0x; col[1] = (_Float16)v0y; col[2] = (_Float16)v0z;
    col[3] = (_Float16)v1x; col[4] = (_Float16)v1y; col[5] = (_Float16)v1z;
#pragma unroll
    for (int j = 0; j < 4; ++j) *(h8*)(X + 1 * 1024 + lane * 32 + j * 8) = *(h8*)(col + j * 8);
    // branch 2: x12 = (v1, v2)
    col[0] = (_Float16)v1x; col[1] = (_Float16)v1y; col[2] = (_Float16)v1z;
    col[3] = (_Float16)v2x; col[4] = (_Float16)v2y; col[5] = (_Float16)v2z;
#pragma unroll
    for (int j = 0; j < 4; ++j) *(h8*)(X + 2 * 1024 + lane * 32 + j * 8) = *(h8*)(col + j * 8);
    // branch 3: x20 = (v2, v0)
    col[0] = (_Float16)v2x; col[1] = (_Float16)v2y; col[2] = (_Float16)v2z;
    col[3] = (_Float16)v0x; col[4] = (_Float16)v0y; col[5] = (_Float16)v0z;
#pragma unroll
    for (int j = 0; j < 4; ++j) *(h8*)(X + 3 * 1024 + lane * 32 + j * 8) = *(h8*)(col + j * 8);
  }
  wave_lds_fence();

  // ---- Layer 1: 4 branches, K=32 (padded), M=64 -> H[br] = relu(W1' x + b1') ----
#pragma unroll
  for (int br = 0; br < 4; ++br) {
    const _Float16* W1 = Wlds + (br == 0 ? 2048 : 0);
    const float*    b1 = Blds + (br == 0 ? 64 : 0);
    h16 vb0 = loadB(X + br * 1024, 32, 0, lane, 0);
    h16 vb1 = loadB(X + br * 1024, 32, 0, lane, 1);
#pragma unroll
    for (int mt = 0; mt < 4; ++mt) {
      h16 a = loadA(W1, 32, mt, 0, lane);
      f8v acc0 = wmma_f16(a, vb0, zero8());
      f8v acc1 = wmma_f16(a, vb1, zero8());
      store_relu_h(H + br * 2048, 64, mt, lane, 0, acc0, b1);
      store_relu_h(H + br * 2048, 64, mt, lane, 1, acc1, b1);
    }
  }
  wave_lds_fence();

  _Float16* S   = X;  // reuse: [32][128] operand for fusion layer
  const int n   = lane & 15;
  const int hi8 = (lane >> 4) * 8;

  // ---- Layer 2 (center): cfeat -> S rows 0..63 ----
#pragma unroll
  for (int mt = 0; mt < 4; ++mt) {
    f8v acc0 = zero8(), acc1 = zero8();
#pragma unroll
    for (int kt = 0; kt < 2; ++kt) {
      h16 a = loadA(Wlds + 8192, 64, mt, kt, lane);
      acc0 = wmma_f16(a, loadB(H, 64, kt, lane, 0), acc0);
      acc1 = wmma_f16(a, loadB(H, 64, kt, lane, 1), acc1);
    }
    int ch0 = mt * 16 + hi8;
    h8 hv0, hv1;
#pragma unroll
    for (int r = 0; r < 8; ++r) {
      hv0[r] = (_Float16)fmaxf(acc0[r] + Blds[192 + ch0 + r], 0.0f);
      hv1[r] = (_Float16)fmaxf(acc1[r] + Blds[192 + ch0 + r], 0.0f);
    }
    *(h8*)(S + n * 128 + ch0) = hv0;
    *(h8*)(S + (16 + n) * 128 + ch0) = hv1;
  }

  // ---- Layer 2 (vertex): sum of 3 relu'd GEMMs -> S rows 64..127 (1/3 folded into sW) ----
#pragma unroll
  for (int mt = 0; mt < 4; ++mt) {
    int ch0 = mt * 16 + hi8;
    float sum0[8], sum1[8];
#pragma unroll
    for (int r = 0; r < 8; ++r) { sum0[r] = 0.0f; sum1[r] = 0.0f; }
#pragma unroll
    for (int br = 1; br <= 3; ++br) {
      f8v acc0 = zero8(), acc1 = zero8();
#pragma unroll
      for (int kt = 0; kt < 2; ++kt) {
        h16 a = loadA(Wlds + 4096, 64, mt, kt, lane);
        acc0 = wmma_f16(a, loadB(H + br * 2048, 64, kt, lane, 0), acc0);
        acc1 = wmma_f16(a, loadB(H + br * 2048, 64, kt, lane, 1), acc1);
      }
#pragma unroll
      for (int r = 0; r < 8; ++r) {
        sum0[r] += fmaxf(acc0[r] + Blds[128 + ch0 + r], 0.0f);
        sum1[r] += fmaxf(acc1[r] + Blds[128 + ch0 + r], 0.0f);
      }
    }
    h8 hv0, hv1;
#pragma unroll
    for (int r = 0; r < 8; ++r) {
      hv0[r] = (_Float16)sum0[r];
      hv1[r] = (_Float16)sum1[r];
    }
    *(h8*)(S + n * 128 + 64 + ch0) = hv0;
    *(h8*)(S + (16 + n) * 128 + 64 + ch0) = hv1;
  }
  wave_lds_fence();

  // ---- Fusion layer: out = relu(sW' [cfeat; vsum] + bs), K=128 ----
  const _Float16* sW = Wlds + 12288;
  float* outp = out + (size_t)b * 64 * FN_ + f0 + n;
#pragma unroll
  for (int mt = 0; mt < 4; ++mt) {
    f8v acc0 = zero8(), acc1 = zero8();
#pragma unroll
    for (int kt = 0; kt < 4; ++kt) {
      h16 a = loadA(sW, 128, mt, kt, lane);
      acc0 = wmma_f16(a, loadB(S, 128, kt, lane, 0), acc0);
      acc1 = wmma_f16(a, loadB(S, 128, kt, lane, 1), acc1);
    }
    int ch0 = mt * 16 + hi8;
#pragma unroll
    for (int r = 0; r < 8; ++r) {
      int ch = ch0 + r;
      outp[(size_t)ch * FN_]      = fmaxf(acc0[r] + Blds[256 + ch], 0.0f);
      outp[(size_t)ch * FN_ + 16] = fmaxf(acc1[r] + Blds[256 + ch], 0.0f);
    }
  }
}

extern "C" void kernel_launch(void* const* d_in, const int* in_sizes, int n_in,
                              void* d_out, int out_size, void* d_ws, size_t ws_size,
                              hipStream_t stream) {
  const float* centers = (const float*)d_in[0];
  const float* verts   = (const float*)d_in[1];
  const int*   faces   = (const int*)d_in[2];

  PrepArgs pa;
  pa.cw1 = (const float*)d_in[3];  pa.cb1 = (const float*)d_in[4];
  pa.c1g = (const float*)d_in[5];  pa.c1bt = (const float*)d_in[6];
  pa.c1m = (const float*)d_in[7];  pa.c1v = (const float*)d_in[8];
  pa.cw2 = (const float*)d_in[9];  pa.cb2 = (const float*)d_in[10];
  pa.c2g = (const float*)d_in[11]; pa.c2bt = (const float*)d_in[12];
  pa.c2m = (const float*)d_in[13]; pa.c2v = (const float*)d_in[14];
  pa.vw1 = (const float*)d_in[15]; pa.vb1 = (const float*)d_in[16];
  pa.v1g = (const float*)d_in[17]; pa.v1bt = (const float*)d_in[18];
  pa.v1m = (const float*)d_in[19]; pa.v1v = (const float*)d_in[20];
  pa.vw2 = (const float*)d_in[21]; pa.vb2 = (const float*)d_in[22];
  pa.v2g = (const float*)d_in[23]; pa.v2bt = (const float*)d_in[24];
  pa.v2m = (const float*)d_in[25]; pa.v2v = (const float*)d_in[26];
  pa.sw  = (const float*)d_in[27]; pa.sb  = (const float*)d_in[28];
  pa.s1g = (const float*)d_in[29]; pa.s1bt = (const float*)d_in[30];
  pa.s1m = (const float*)d_in[31]; pa.s1v = (const float*)d_in[32];

  _Float16* wH = (_Float16*)d_ws;
  float*    bF = (float*)((char*)d_ws + 40960);
  hipLaunchKernelGGL(prep_kernel, dim3(1), dim3(64), 0, stream, pa, wH, bF);

  const int tiles  = (BN_ * FN_) / 32;  // 25000 macro-tiles
  const int blocks = tiles / 4;         // 6250 (4 waves/block, 1 macro-tile/wave)
  const size_t shmem = 42240 + 4 * 24576;  // 140544 B
  hipLaunchKernelGGL(spatial_desc_main, dim3(blocks), dim3(128), shmem, stream,
                     centers, verts, faces, (const uint4*)d_ws, (float*)d_out);
}
